// Evolution_85375359910669
// MI455X (gfx1250) — matrix operand order repaired and checked by reading, not compiled, not run on hardware
//
#include <hip/hip_runtime.h>
#include <stddef.h>

typedef __attribute__((ext_vector_type(16))) __bf16 v16bf;
typedef __attribute__((ext_vector_type(8)))  float  v8f;
typedef __attribute__((ext_vector_type(4)))  unsigned u32x4;
typedef unsigned short u16;

// ---------------- LDS layout (dynamic, per workgroup) ----------------
// states: 8 * [128 pt][128 ch] bf16 = 262144
#define LDS_STATES   0
#define LDS_GLOBU    262144      // 256 * u32 (max-reduce scratch)
#define LDS_GLOBBF   263168      // 256 * bf16 (global feature, broadcast B operand)
#define LDS_EPB      263680      // 256 * f32  (per-layer bias)
#define LDS_EPS      264704      // 128 * f32  (per-layer BN scale)
#define LDS_EPT      265216      // 128 * f32  (per-layer BN shift)
#define LDS_H1       265728      // [128 pt][64 ch] bf16 = 16384
#define LDS_TOTAL    282112      // < 320 KB WGP LDS

// ---------------- helpers ----------------
__device__ __forceinline__ u16 f2bfu(float f) {
  unsigned u = __float_as_uint(f);
  unsigned r = u + 0x7FFFu + ((u >> 16) & 1u);   // RNE
  return (u16)(r >> 16);
}
__device__ __forceinline__ float bf2f(u16 h) { return __uint_as_float(((unsigned)h) << 16); }
__device__ __forceinline__ unsigned fmapu(float f) {   // order-preserving float->uint
  unsigned u = __float_as_uint(f);
  return (u & 0x80000000u) ? ~u : (u | 0x80000000u);
}
__device__ __forceinline__ float funmap(unsigned u) {
  unsigned v = (u & 0x80000000u) ? (u & 0x7FFFFFFFu) : ~u;
  return __uint_as_float(v);
}

union FragAB { v16bf v; u32x4 q[2]; };
union FragC  { v8f   v; float f[8]; };
union Pack8  { u32x4 q; u16   us[8]; };

__device__ __forceinline__ void wmma_bf16(FragC& c, const FragAB& a, const FragAB& b) {
  c.v = __builtin_amdgcn_wmma_f32_16x16x32_bf16(false, a.v, false, b.v, (short)0, c.v, false, false);
}

// ---------------- weight fragment packing ----------------
// Packs W (conv OIK [O][I][taps] or linear [O][I], taps==1) into WMMA A-fragments:
// frag = (t*KCH + kc)*MT + mt, 1024 B each: lane holds 32 contiguous bytes (16 bf16)
// row m = lane%16, k = 16*(lane/16) + e, contiguous order (B loads match). In-channels padded to Ipad with 0.
__global__ __launch_bounds__(256)
void pack_w_kernel(const float* __restrict__ w, u16* __restrict__ dst,
                   int O, int I, int Ipad, int taps, int total) {
  int gid = blockIdx.x * 256 + threadIdx.x;
  if (gid >= total) return;
  int e    = gid & 15;
  int lane = (gid >> 4) & 31;
  int frag = gid >> 9;
  int MT = O >> 4, KCH = Ipad >> 5;
  int mt = frag % MT;
  int kc = (frag / MT) % KCH;
  int t  = frag / (MT * KCH);
  int outc = mt * 16 + (lane & 15);
  int half = lane >> 4, vv = e >> 1;
  int kof = half * 16 + e; (void)vv;
  int in = kc * 32 + kof;
  float val = (in < I) ? w[((size_t)outc * I + in) * taps + t] : 0.0f;
  dst[(size_t)frag * 512 + lane * 16 + e] = f2bfu(val);
}

// ---------------- per-iteration input build (poly prep + bilinear gather) ----------------
__global__ __launch_bounds__(128)
void build_inp_kernel(const float* __restrict__ feat, const float* __restrict__ img_polys,
                      const float* __restrict__ can_polys, const int* __restrict__ ind,
                      const float* __restrict__ prev_py, float* __restrict__ poly_cur,
                      u16* __restrict__ inp, int iter) {
  __shared__ float red[256];
  int poly = blockIdx.x, pt = threadIdx.x;
  size_t pidx = ((size_t)poly * 128 + pt) * 2;
  float px, py, cx, cy;
  if (iter == 0) {
    px = fminf(fmaxf(img_polys[pidx + 0], 0.f), 127.f);
    py = fminf(fmaxf(img_polys[pidx + 1], 0.f), 127.f);
    cx = can_polys[pidx + 0]; cy = can_polys[pidx + 1];
  } else {
    px = prev_py[pidx + 0] * 0.25f;
    py = prev_py[pidx + 1] * 0.25f;
    red[pt] = px; red[128 + pt] = py;
    __syncthreads();
    for (int s = 64; s > 0; s >>= 1) {
      if (pt < s) {
        red[pt]       = fminf(red[pt],       red[pt + s]);
        red[128 + pt] = fminf(red[128 + pt], red[128 + pt + s]);
      }
      __syncthreads();
    }
    cx = px - red[0]; cy = py - red[128];
  }
  poly_cur[pidx + 0] = px; poly_cur[pidx + 1] = py;

  // grid_sample, align_corners=False, zeros padding; for H=W=128: x = px - 0.5
  float x = px - 0.5f, y = py - 0.5f;
  float x0f = floorf(x), y0f = floorf(y);
  int x0 = (int)x0f, y0 = (int)y0f;
  float wx = x - x0f, wy = y - y0f;
  float w00 = (1.f - wy) * (1.f - wx), w01 = (1.f - wy) * wx;
  float w10 = wy * (1.f - wx),         w11 = wy * wx;
  bool vx0 = (x0 >= 0) & (x0 < 128), vx1 = (x0 + 1 >= 0) & (x0 + 1 < 128);
  bool vy0 = (y0 >= 0) & (y0 < 128), vy1 = (y0 + 1 >= 0) & (y0 + 1 < 128);
  int b = ind[poly];
  const float* fb = feat + (size_t)b * 64 * 128 * 128;
  u16* dst = inp + ((size_t)poly * 128 + pt) * 96;
  int r0 = y0 * 128, r1 = r0 + 128;
  for (int c = 0; c < 64; ++c) {
    const float* fc = fb + (size_t)c * 16384;
    float v = 0.f;
    if (vy0 & vx0) v += w00 * fc[r0 + x0];
    if (vy0 & vx1) v += w01 * fc[r0 + x0 + 1];
    if (vy1 & vx0) v += w10 * fc[r1 + x0];
    if (vy1 & vx1) v += w11 * fc[r1 + x0 + 1];
    dst[c] = f2bfu(v);
  }
  dst[64] = f2bfu(cx * 4.0f);   // c_it * RO
  dst[65] = f2bfu(cy * 4.0f);
  for (int c = 66; c < 96; ++c) dst[c] = 0;   // K padding
}

// ---------------- circular dilated conv layer as tap-wise WMMA GEMM ----------------
// Y[128 out][128 pt] = sum_t W_t[128 x KCH*32] @ X[.][pt + (t-4)*dil mod 128]
// 8 waves: wave w owns out-channel tile mt=w; all 8 point tiles live in accumulators
// so each A (weight) fragment load feeds 8 back-to-back WMMAs.
template<int KCH, bool FROM_GLOBAL, bool RESIDUAL>
__device__ __forceinline__ void conv_layer(char* smem, const char* __restrict__ wpack,
                                           const char* __restrict__ inp_g,
                                           int srcOff, int dstOff, int dil,
                                           int poly, int wave, int lane) {
  const float* ep_b = (const float*)(smem + LDS_EPB);
  const float* ep_s = (const float*)(smem + LDS_EPS);
  const float* ep_t = (const float*)(smem + LDS_EPT);
  const int mt = wave;
  const int nlo = lane & 15;
  const int half16 = (lane >> 4) << 4;
  FragC acc[8];
#pragma unroll
  for (int nt = 0; nt < 8; ++nt)
#pragma unroll
    for (int j = 0; j < 8; ++j) acc[nt].f[j] = 0.0f;

  for (int t = 0; t < 9; ++t) {
    const int sh = (t - 4) * dil + 128;
#pragma unroll
    for (int kc = 0; kc < KCH; ++kc) {
      FragAB A;
      const u32x4* ap = (const u32x4*)(wpack + (size_t)((t * KCH + kc) * 8 + mt) * 1024 + lane * 32);
      A.q[0] = ap[0]; A.q[1] = ap[1];
      const int ch = kc * 32 + half16;
#pragma unroll
      for (int nt = 0; nt < 8; ++nt) {
        int pt = (nt * 16 + nlo + sh) & 127;   // circular shift
        FragAB B;
        if (FROM_GLOBAL) {
          const u32x4* bp = (const u32x4*)(inp_g + (((size_t)poly * 128 + pt) * 96 + ch) * 2);
          B.q[0] = bp[0]; B.q[1] = bp[1];
        } else {
          const u32x4* bp = (const u32x4*)(smem + srcOff + ((pt * 128 + ch) << 1));
          B.q[0] = bp[0]; B.q[1] = bp[1];
        }
        wmma_bf16(acc[nt], A, B);
      }
    }
  }
  // epilogue: bias -> relu -> BN(eval) -> (+residual) -> bf16 store (8 consecutive ch / lane)
  const int ch0 = (mt << 4) + ((lane >> 4) << 3);
#pragma unroll
  for (int nt = 0; nt < 8; ++nt) {
    int ptd = nt * 16 + nlo;
    float r8[8];
    if (RESIDUAL) {
      Pack8 r;
      r.q = *(const u32x4*)(smem + srcOff + ((ptd * 128 + ch0) << 1));
#pragma unroll
      for (int j = 0; j < 8; ++j) r8[j] = bf2f(r.us[j]);
    }
    Pack8 o;
#pragma unroll
    for (int j = 0; j < 8; ++j) {
      float v = acc[nt].f[j] + ep_b[ch0 + j];
      v = fmaxf(v, 0.0f);
      v = v * ep_s[ch0 + j] + ep_t[ch0 + j];
      if (RESIDUAL) v += r8[j];
      o.us[j] = f2bfu(v);
    }
    *(u32x4*)(smem + dstOff + ((ptd * 128 + ch0) << 1)) = o.q;
  }
}

// ---------------- snake forward: one workgroup per polygon ----------------
struct SnakeArgs {
  const char* wp_head;
  const char* wp_res[7];
  const char* wp_fus;
  const char* wp_p0;
  const char* wp_p1;
  const float* head_b; const float* head_g; const float* head_bt;
  const float* res_b[7]; const float* res_g[7]; const float* res_bt[7];
  const float* fus_b; const float* p0_b; const float* p1_b;
  const float* p2_w;  const float* p2_b;
  const u16* inp;        // [N][128][96] bf16
  const float* poly_cur; // [N][128][2] f32
  u16* h0;               // [N][128][256] bf16 scratch
  float* out;            // [N][128][2] f32 (slice of d_out)
};

__global__ __launch_bounds__(256)
void snake_kernel(SnakeArgs a) {
  extern __shared__ char smem[];
  const int tid = threadIdx.x;
  const int wave = tid >> 5, lane = tid & 31;
  const int poly = blockIdx.x;
  float* ep_b = (float*)(smem + LDS_EPB);
  float* ep_s = (float*)(smem + LDS_EPS);
  float* ep_t = (float*)(smem + LDS_EPT);
  unsigned* glob_u = (unsigned*)(smem + LDS_GLOBU);
  u16* glob_bf = (u16*)(smem + LDS_GLOBBF);
  const int half16 = (lane >> 4) << 4;
  const int nlo = lane & 15;
  const int DILS[7] = {1, 1, 1, 2, 2, 4, 4};

  // ---- head conv (input from global bf16, 96 padded ch) -> state0 ----
  if (tid < 128) {
    ep_b[tid] = a.head_b[tid];
    ep_s[tid] = a.head_g[tid] * rsqrtf(1.0f + 1e-5f);
    ep_t[tid] = a.head_bt[tid];
  }
  __syncthreads();
  conv_layer<3, true, false>(smem, a.wp_head, (const char*)a.inp,
                             0, LDS_STATES, 1, poly, wave, lane);
  __syncthreads();

  // ---- 7 residual dilated conv blocks: state[i] -> state[i+1] ----
  for (int i = 0; i < 7; ++i) {
    if (tid < 128) {
      ep_b[tid] = a.res_b[i][tid];
      ep_s[tid] = a.res_g[i][tid] * rsqrtf(1.0f + 1e-5f);
      ep_t[tid] = a.res_bt[i][tid];
    }
    __syncthreads();
    conv_layer<4, false, true>(smem, a.wp_res[i], nullptr,
                               LDS_STATES + i * 32768, LDS_STATES + (i + 1) * 32768,
                               DILS[i], poly, wave, lane);
    __syncthreads();
  }

  // ---- fusion: 256 x 1024 GEMM over concat states, max-pool over points ----
  // wave handles mt = wave*2 + pass; 8 nt accumulators share each A fragment.
  if (tid < 256) { ep_b[tid] = a.fus_b[tid]; glob_u[tid] = 0u; }
  __syncthreads();
  for (int pass = 0; pass < 2; ++pass) {
    const int mt = wave * 2 + pass;
    FragC acc[8];
#pragma unroll
    for (int nt = 0; nt < 8; ++nt)
#pragma unroll
      for (int j = 0; j < 8; ++j) acc[nt].f[j] = 0.0f;
    for (int kc = 0; kc < 32; ++kc) {
      FragAB A;
      const u32x4* ap = (const u32x4*)(a.wp_fus + (size_t)(kc * 16 + mt) * 1024 + lane * 32);
      A.q[0] = ap[0]; A.q[1] = ap[1];
      const int s = kc >> 2, chs = (kc & 3) * 32 + half16;
      const char* bbase = smem + LDS_STATES + s * 32768 + (chs << 1);
#pragma unroll
      for (int nt = 0; nt < 8; ++nt) {
        int pt = nt * 16 + nlo;
        const u32x4* bp = (const u32x4*)(bbase + (pt << 8));
        FragAB B; B.q[0] = bp[0]; B.q[1] = bp[1];
        wmma_bf16(acc[nt], A, B);
      }
    }
    const int ch0 = (mt << 4) + ((lane >> 4) << 3);
#pragma unroll
    for (int nt = 0; nt < 8; ++nt)
#pragma unroll
      for (int j = 0; j < 8; ++j) {
        float v = acc[nt].f[j] + ep_b[ch0 + j];
        atomicMax(&glob_u[ch0 + j], fmapu(v));
      }
  }
  __syncthreads();
  if (tid < 256) {
    glob_bf[tid] = f2bfu(funmap(glob_u[tid]));
    ep_b[tid] = a.p0_b[tid];
  }
  __syncthreads();

  // ---- p0: 256 x 1280 GEMM over [glob(256); states(1024)] -> relu -> h0 (global bf16) ----
  for (int pass = 0; pass < 2; ++pass) {
    const int mt = wave * 2 + pass;
    FragC acc[8];
#pragma unroll
    for (int nt = 0; nt < 8; ++nt)
#pragma unroll
      for (int j = 0; j < 8; ++j) acc[nt].f[j] = 0.0f;
    for (int kc = 0; kc < 40; ++kc) {
      FragAB A;
      const u32x4* ap = (const u32x4*)(a.wp_p0 + (size_t)(kc * 16 + mt) * 1024 + lane * 32);
      A.q[0] = ap[0]; A.q[1] = ap[1];
      if (kc < 8) {   // broadcast global-feature rows (point-invariant B)
        const u32x4* bp = (const u32x4*)(smem + LDS_GLOBBF + ((kc * 32 + half16) << 1));
        FragAB B; B.q[0] = bp[0]; B.q[1] = bp[1];
#pragma unroll
        for (int nt = 0; nt < 8; ++nt) wmma_bf16(acc[nt], A, B);
      } else {
        const int k2 = kc - 8, s = k2 >> 2, chs = (k2 & 3) * 32 + half16;
        const char* bbase = smem + LDS_STATES + s * 32768 + (chs << 1);
#pragma unroll
        for (int nt = 0; nt < 8; ++nt) {
          int pt = nt * 16 + nlo;
          const u32x4* bp = (const u32x4*)(bbase + (pt << 8));
          FragAB B; B.q[0] = bp[0]; B.q[1] = bp[1];
          wmma_bf16(acc[nt], A, B);
        }
      }
    }
    const int ch0 = (mt << 4) + ((lane >> 4) << 3);
#pragma unroll
    for (int nt = 0; nt < 8; ++nt) {
      int pt = nt * 16 + nlo;
      Pack8 o;
#pragma unroll
      for (int j = 0; j < 8; ++j) o.us[j] = f2bfu(fmaxf(acc[nt].f[j] + ep_b[ch0 + j], 0.0f));
      *(u32x4*)((char*)a.h0 + ((((size_t)poly * 128 + pt) * 256 + ch0) << 1)) = o.q;
    }
  }
  __threadfence();
  __syncthreads();
  if (tid < 64) ep_b[tid] = a.p1_b[tid];
  __syncthreads();

  // ---- p1: 64 x 256 GEMM over h0 -> relu -> h1 (LDS bf16) ----
  // wave -> (mt = wave/2, 4 nt tiles); A fragment shared by 4 WMMAs.
  {
    const int mt = wave >> 1;
    const int ntb = (wave & 1) * 4;
    FragC acc[4];
#pragma unroll
    for (int i = 0; i < 4; ++i)
#pragma unroll
      for (int j = 0; j < 8; ++j) acc[i].f[j] = 0.0f;
    for (int kc = 0; kc < 8; ++kc) {
      FragAB A;
      const u32x4* ap = (const u32x4*)(a.wp_p1 + (size_t)(kc * 4 + mt) * 1024 + lane * 32);
      A.q[0] = ap[0]; A.q[1] = ap[1];
#pragma unroll
      for (int i = 0; i < 4; ++i) {
        int pt = (ntb + i) * 16 + nlo;
        const u32x4* bp = (const u32x4*)((const char*)a.h0 +
            ((((size_t)poly * 128 + pt) * 256 + kc * 32 + half16) << 1));
        FragAB B; B.q[0] = bp[0]; B.q[1] = bp[1];
        wmma_bf16(acc[i], A, B);
      }
    }
    const int ch0 = (mt << 4) + ((lane >> 4) << 3);
#pragma unroll
    for (int i = 0; i < 4; ++i) {
      int pt = (ntb + i) * 16 + nlo;
      Pack8 o;
#pragma unroll
      for (int j = 0; j < 8; ++j) o.us[j] = f2bfu(fmaxf(acc[i].f[j] + ep_b[ch0 + j], 0.0f));
      *(u32x4*)(smem + LDS_H1 + ((pt * 64 + ch0) << 1)) = o.q;
    }
  }
  __syncthreads();

  // ---- p2 (2 x 64) scalar dot + poly update: py = i_it*RO + off ----
  {
    int pt = tid & 127, comp = tid >> 7;
    const u16* h1 = (const u16*)(smem + LDS_H1);
    float acc = a.p2_b[comp];
    for (int k = 0; k < 64; ++k)
      acc += bf2f(h1[pt * 64 + k]) * a.p2_w[comp * 64 + k];
    float base = a.poly_cur[((size_t)poly * 128 + pt) * 2 + comp];
    a.out[((size_t)poly * 128 + pt) * 2 + comp] = base * 4.0f + acc;
  }
}

// ---------------- host launch ----------------
extern "C" void kernel_launch(void* const* d_in, const int* in_sizes, int n_in,
                              void* d_out, int out_size, void* d_ws, size_t ws_size,
                              hipStream_t stream) {
  (void)in_sizes; (void)out_size;
  const float* feat = (const float*)d_in[0];
  const float* imgp = (const float*)d_in[1];
  const float* canp = (const float*)d_in[2];
  const int*   ind  = (const int*)d_in[n_in - 1];
  // params flattened in insertion order: 40 leaves per snake starting at index 3
  auto sp = [&](int s, int k) { return (const float*)d_in[3 + s * 40 + k]; };

  char* ws = (char*)d_ws;
  size_t off = 0;
  auto alloc = [&](size_t sz) { size_t o = off; off += (sz + 255) & ~(size_t)255; return o; };
  size_t inp_o  = alloc((size_t)256 * 128 * 96 * 2);
  size_t poly_o = alloc((size_t)256 * 128 * 2 * 4);
  size_t h0_o   = alloc((size_t)256 * 128 * 256 * 2);
  size_t wp_head[3], wp_res[3][7], wp_fus[3], wp_p0[3], wp_p1[3];
  for (int s = 0; s < 3; ++s) {
    wp_head[s] = alloc((size_t)9 * 3 * 8 * 1024);
    for (int i = 0; i < 7; ++i) wp_res[s][i] = alloc((size_t)9 * 4 * 8 * 1024);
    wp_fus[s] = alloc((size_t)32 * 16 * 1024);
    wp_p0[s]  = alloc((size_t)40 * 16 * 1024);
    wp_p1[s]  = alloc((size_t)8 * 4 * 1024);
  }
  if (ws_size < off) return;   // scratch too small: bail deterministically

  auto pack = [&](const float* w, size_t dsto, int O, int I, int Ipad, int taps) {
    int total = taps * (Ipad >> 5) * (O >> 4) * 512;
    pack_w_kernel<<<(total + 255) / 256, 256, 0, stream>>>(
        w, (u16*)(ws + dsto), O, I, Ipad, taps, total);
  };
  for (int s = 0; s < 3; ++s) {
    pack(sp(s, 0), wp_head[s], 128, 66, 96, 9);
    for (int i = 0; i < 7; ++i) pack(sp(s, 4 + 4 * i), wp_res[s][i], 128, 128, 128, 9);
    pack(sp(s, 32), wp_fus[s], 256, 1024, 1024, 1);
    pack(sp(s, 34), wp_p0[s], 256, 1280, 1280, 1);
    pack(sp(s, 36), wp_p1[s], 64, 256, 256, 1);
  }

  float* out = (float*)d_out;
  for (int it = 0; it < 3; ++it) {
    const float* prev = (it == 0) ? (const float*)nullptr : out + (size_t)(it - 1) * 256 * 128 * 2;
    build_inp_kernel<<<256, 128, 0, stream>>>(feat, imgp, canp, ind, prev,
        (float*)(ws + poly_o), (u16*)(ws + inp_o), it);

    SnakeArgs A;
    A.wp_head = ws + wp_head[it];
    for (int i = 0; i < 7; ++i) A.wp_res[i] = ws + wp_res[it][i];
    A.wp_fus = ws + wp_fus[it];
    A.wp_p0  = ws + wp_p0[it];
    A.wp_p1  = ws + wp_p1[it];
    A.head_b = sp(it, 1); A.head_g = sp(it, 2); A.head_bt = sp(it, 3);
    for (int i = 0; i < 7; ++i) {
      A.res_b[i] = sp(it, 5 + 4 * i); A.res_g[i] = sp(it, 6 + 4 * i); A.res_bt[i] = sp(it, 7 + 4 * i);
    }
    A.fus_b = sp(it, 33); A.p0_b = sp(it, 35); A.p1_b = sp(it, 37);
    A.p2_w = sp(it, 38);  A.p2_b = sp(it, 39);
    A.inp = (const u16*)(ws + inp_o);
    A.poly_cur = (const float*)(ws + poly_o);
    A.h0 = (u16*)(ws + h0_o);
    A.out = out + (size_t)it * 256 * 128 * 2;
    snake_kernel<<<256, 256, LDS_TOTAL, stream>>>(A);
  }
}
